// Horizon_AttentionPooling_23776938951107
// MI455X (gfx1250) — compile-verified, matrix operand order checked by reading, not run on hardware
//
#include <hip/hip_runtime.h>
#include <stdint.h>

// Horizon attention pooling, MI455X (gfx1250).
// Key insight: same_seg masking makes attention block-diagonal over 64
// contiguous 32-row segments -> only 8MB of the 512MB `f` tensor is live.
// Roofline: ~10MB moved, ~20 MFLOP -> pure bandwidth/launch bound; use
// TDM (tensor_load_to_lds) for the strided 128KB f-block per segment and
// WMMA for both GEMMs.

#define T 2048
#define NUM_SEG 64
#define SEG 32
#define F_DIM 32
#define H_DIM 128
#define BLK 256

typedef _Float16 v16h __attribute__((ext_vector_type(16)));
typedef float    v8f  __attribute__((ext_vector_type(8)));
typedef float    v2f  __attribute__((ext_vector_type(2)));
typedef unsigned int v4u __attribute__((ext_vector_type(4)));
typedef int      v8i  __attribute__((ext_vector_type(8)));
typedef int      v4i  __attribute__((ext_vector_type(4)));

// ---------------------------------------------------------------------------
// Kernel 1: Wh = h @ W_w^T + W_b   (2048x128 @ 128x32)
// One wave per 16x16 C tile; K swept in steps of 4 with v_wmma_f32_16x16x4_f32.
// ---------------------------------------------------------------------------
__global__ __launch_bounds__(64) void wh_kernel(const float* __restrict__ h,
                                                const float* __restrict__ Ww,
                                                const float* __restrict__ Wb,
                                                float* __restrict__ Wh) {
  const int mtile = blockIdx.x;        // 0..127 (rows of h, 16 at a time)
  const int wave  = threadIdx.x >> 5;  // 0..1  -> N tile (cols of Wh)
  const int lane  = threadIdx.x & 31;
  const int half  = lane >> 4;
  const int l15   = lane & 15;

#if __has_builtin(__builtin_amdgcn_wmma_f32_16x16x4_f32)
  // A (16x4 f32): lane m=l15, VGPR0/1 = K = 2*half, 2*half+1
  // B (4x16 f32): lane n=l15, same K striping (mirror layout)
  const int m = mtile * 16 + l15;
  const int n = wave * 16 + l15;
  v8f c = {};
  for (int k0 = 0; k0 < H_DIM; k0 += 4) {
    v2f a, b;
    a.x = h[m * H_DIM + k0 + 2 * half];
    a.y = h[m * H_DIM + k0 + 2 * half + 1];
    b.x = Ww[n * H_DIM + k0 + 2 * half];      // B[k][n] = W_w[n][k]
    b.y = Ww[n * H_DIM + k0 + 2 * half + 1];
    c = __builtin_amdgcn_wmma_f32_16x16x4_f32(false, a, false, b,
                                              (short)0, c, false, false);
  }
  const float bias = Wb[wave * 16 + l15];
#pragma unroll
  for (int r = 0; r < 8; ++r) {
    const int row = mtile * 16 + r + 8 * half;   // C: M = r + 8*half, N = l15
    Wh[row * F_DIM + wave * 16 + l15] = c[r] + bias;
  }
#else
  // VALU fallback: 64 threads, 4 threads/row, 8 outputs each.
  const int row = mtile * 16 + (threadIdx.x >> 2);
  for (int n = (threadIdx.x & 3); n < F_DIM; n += 4) {
    float acc = Wb[n];
    for (int k = 0; k < H_DIM; ++k) acc += h[row * H_DIM + k] * Ww[n * H_DIM + k];
    Wh[row * F_DIM + n] = acc;
  }
  (void)wave; (void)lane; (void)half; (void)l15;
#endif
}

// ---------------------------------------------------------------------------
// Kernel 2: one workgroup per segment (64 blocks x 256 threads = 8 waves).
//   TDM-load f[i, seg, :] block (128KB, 2D strided) -> LDS
//   logits + mask -> softmax -> S_seg = attn @ h_seg via v_wmma_f32_16x16x32_f16
// ---------------------------------------------------------------------------
__global__ __launch_bounds__(BLK) void seg_attn_kernel(
    const float* __restrict__ f, const float* __restrict__ h,
    const float* __restrict__ features, const float* __restrict__ hor,
    const long long* __restrict__ sub_batches,
    const float* __restrict__ Wh, float* __restrict__ S) {

  __shared__ __align__(16) float    sh_f[SEG * SEG * F_DIM];  // 128 KB
  __shared__ __align__(16) float    sh_wh[SEG * F_DIM];       //   4 KB
  __shared__ __align__(16) _Float16 sh_h16[SEG * H_DIM];      //   8 KB
  __shared__ __align__(16) float    sh_logit[SEG * SEG];      //   4 KB
  __shared__ __align__(16) _Float16 sh_attn[SEG * SEG];       //   2 KB

  const int s    = blockIdx.x;
  const int row0 = (int)sub_batches[2 * s];
  const int segsz = (int)sub_batches[2 * s + 1] - row0;  // 32 for this setup
  const int tid  = threadIdx.x;

  // ---- stage f block: rows i in [row0,row0+32), cols j in seg, all F_DIM ----
  const uint64_t gbase =
      (uint64_t)(uintptr_t)(f + (size_t)row0 * T * F_DIM + (size_t)row0 * F_DIM);
#if __has_builtin(__builtin_amdgcn_tensor_load_to_lds)
  if (tid < 32) {  // single wave issues the DMA
    const unsigned lds_addr = (unsigned)(uintptr_t)&sh_f[0];  // flat low 32 = LDS offset
    v4u g0;
    g0.x = 1u;                                       // count=1, user descriptor
    g0.y = lds_addr;                                 // lds_addr
    g0.z = (unsigned)(gbase & 0xFFFFFFFFu);          // global_addr[31:0]
    g0.w = ((unsigned)((gbase >> 32) & 0x1FFFFFFu))  // global_addr[56:32]
           | (2u << 30);                             // type=2 ("image")
    v8i g1;
    g1[0] = 0x00020000;                              // data_size=2 (4B), no flags
    g1[1] = (int)(((unsigned)(T * F_DIM) & 0xFFFFu) << 16);       // dim0 lo16 (=0)
    g1[2] = (int)((((unsigned)T & 0xFFFFu) << 16) |               // tensor_dim1 lo16
                  (((unsigned)(T * F_DIM)) >> 16));               // tensor_dim0 hi16
    g1[3] = (int)(((unsigned)(SEG * F_DIM) & 0xFFFFu) << 16);     // tile_dim0=1024
    g1[4] = SEG;                                                  // tile_dim1=32
    g1[5] = T * F_DIM;                                            // dim0_stride=65536
    g1[6] = 0;
    g1[7] = 0;
    v4i gz = {};
#if defined(__clang_major__) && __clang_major__ >= 23
    v8i g1z = {};
    __builtin_amdgcn_tensor_load_to_lds(g0, g1, gz, gz, g1z, 0);
#else
    __builtin_amdgcn_tensor_load_to_lds(g0, g1, gz, gz, 0);
#endif
    __builtin_amdgcn_s_wait_tensorcnt(0);
  }
#else
  // Fallback: cooperative strided copy, float4 granularity.
  for (int idx4 = tid; idx4 < SEG * SEG * F_DIM / 4; idx4 += BLK) {
    const int idx = idx4 * 4;
    const int i = idx >> 10;          // row within segment
    const int off = idx & 1023;       // offset within 1024-float row slab
    *(float4*)&sh_f[idx] =
        *(const float4*)&f[((size_t)(row0 + i) * T + row0) * F_DIM + off];
  }
#endif

  // ---- stage h_seg as f16 (contiguous 16KB) and Wh_seg (contiguous 4KB) ----
  for (int idx = tid; idx < SEG * H_DIM; idx += BLK)
    sh_h16[idx] = (_Float16)h[(size_t)row0 * H_DIM + idx];
  for (int idx = tid; idx < SEG * F_DIM; idx += BLK)
    sh_wh[idx] = Wh[(size_t)row0 * F_DIM + idx];
  __syncthreads();

  // ---- logits[i][j] = f[i,j,:]·Wh[j,:], masked ----
  for (int p = tid; p < SEG * SEG; p += BLK) {
    const int i = p >> 5, j = p & 31;
    const int gi = row0 + i, gj = row0 + j;
    const float4* fa = (const float4*)&sh_f[(i * SEG + j) * F_DIM];
    const float4* wa = (const float4*)&sh_wh[j * F_DIM];
    float acc = 0.f;
#pragma unroll
    for (int k = 0; k < F_DIM / 4; ++k) {
      const float4 a = fa[k], b = wa[k];
      acc += a.x * b.x + a.y * b.y + a.z * b.z + a.w * b.w;
    }
    const float dist = features[((size_t)gi * T + gj) * 4 + 0];
    const float hb = hor[(size_t)gi * T + gj];
    const bool bad = (hb < 0.f) | (dist > 10.f) | (gi == gj);
    sh_logit[p] = bad ? -1000.f : acc;
  }
  __syncthreads();

  // ---- per-row softmax over the 32 in-segment entries ----
  if (tid < SEG) {
    const int i = tid;
    float mx = -3.402823466e38f;
#pragma unroll
    for (int j = 0; j < SEG; ++j) mx = fmaxf(mx, sh_logit[i * SEG + j]);
    float sum = 0.f;
#pragma unroll
    for (int j = 0; j < SEG; ++j) sum += __expf(sh_logit[i * SEG + j] - mx);
    const float inv = 1.f / sum;
#pragma unroll
    for (int j = 0; j < SEG; ++j)
      sh_attn[i * SEG + j] = (_Float16)(__expf(sh_logit[i * SEG + j] - mx) * inv);
  }
  __syncthreads();

  // ---- S_seg = attn(32x32) @ h_seg(32x128) via v_wmma_f32_16x16x32_f16 ----
  // 16 C tiles (2 M x 8 N); each of the 8 waves computes 2 tiles. EXEC all-1s.
  const int wave = tid >> 5, lane = tid & 31;
  const int half = lane >> 4, l15 = lane & 15;
  const int mt = wave >> 2;            // 0..1
  const int ntb = (wave & 3) * 2;      // 0,2,4,6

  v16h a;  // A 16x32 f16: lane m=l15 (+half selects K-octet), ISA 16-bit A layout
#pragma unroll
  for (int v = 0; v < 8; ++v) {
    const int kb = ((v < 4) ? 0 : 16) + 8 * half + (v & 3) * 2;
    a[2 * v]     = sh_attn[(mt * 16 + l15) * SEG + kb];
    a[2 * v + 1] = sh_attn[(mt * 16 + l15) * SEG + kb + 1];
  }
  const bool keep = (segsz > 1);  // reference zeroes rows of size-1 segments
#pragma unroll
  for (int t = 0; t < 2; ++t) {
    const int nt = ntb + t;
    v16h b;  // B 32x16 f16: lane n=l15, mirrored K striping
#pragma unroll
    for (int v = 0; v < 8; ++v) {
      const int kb = ((v < 4) ? 0 : 16) + 8 * half + (v & 3) * 2;
      b[2 * v]     = sh_h16[kb * H_DIM + nt * 16 + l15];
      b[2 * v + 1] = sh_h16[(kb + 1) * H_DIM + nt * 16 + l15];
    }
    v8f c = {};
    c = __builtin_amdgcn_wmma_f32_16x16x32_f16(false, a, false, b,
                                               (short)0, c, false, false);
#pragma unroll
    for (int r = 0; r < 8; ++r) {
      const int gi = row0 + mt * 16 + r + 8 * half;  // C: M = r + 8*half, N = l15
      S[(size_t)gi * H_DIM + nt * 16 + l15] = keep ? c[r] : 0.f;
    }
  }
}

// ---------------------------------------------------------------------------
extern "C" void kernel_launch(void* const* d_in, const int* in_sizes, int n_in,
                              void* d_out, int out_size, void* d_ws, size_t ws_size,
                              hipStream_t stream) {
  (void)in_sizes; (void)n_in; (void)out_size; (void)ws_size;
  const float* f    = (const float*)d_in[0];
  const float* h    = (const float*)d_in[1];
  const float* feat = (const float*)d_in[2];
  const float* hor  = (const float*)d_in[3];
  const float* Ww   = (const float*)d_in[4];
  const float* Wb   = (const float*)d_in[5];
  const long long* sb = (const long long*)d_in[6];
  float* S  = (float*)d_out;
  float* Wh = (float*)d_ws;  // T*F_DIM fp32 = 256 KB scratch

  wh_kernel<<<T / 16, 64, 0, stream>>>(h, Ww, Wb, Wh);
  seg_attn_kernel<<<NUM_SEG, BLK, 0, stream>>>(f, h, feat, hor, sb, Wh, S);
}
